// SelsaHead_31490700214886
// MI455X (gfx1250) — matrix operand order repaired and checked by reading, not compiled, not run on hardware
//
#include <hip/hip_runtime.h>
#include <hip/hip_bf16.h>

// ---------------------------------------------------------------------------
// SelsaHead for MI455X (gfx1250): all matmuls via v_wmma_f32_16x16x32_f16.
// f32 inputs converted to f16 in-register while staging into LDS.
// K-loop is double-buffered (LDS x2 + register prefetch) so global loads for
// chunk k+1 are in flight while WMMAs for chunk k execute — required to get
// near the 23.3 TB/s floor on the memory-bound fc1 (308 MB compulsory).
// ---------------------------------------------------------------------------

typedef __attribute__((ext_vector_type(16))) _Float16 v16h;
typedef __attribute__((ext_vector_type(8)))  _Float16 v8h;
typedef __attribute__((ext_vector_type(4)))  _Float16 v4h;
typedef __attribute__((ext_vector_type(8)))  float    v8f;
typedef __attribute__((ext_vector_type(4)))  float    v4f;

#define NPROP 512
#define DDIM  1024
#define FLATK 50176
#define ATTN_SCALE 0.03125f   // 1/sqrt(1024)

// ---------------------------------------------------------------------------
// Tiled WMMA GEMM:  C[M,N] = alpha * (A[M,K] x B') + bias
//   BT=true : B stored [N,K] row-major (weight layout, B' = B^T; also Q.K^T)
//   BT=false: B stored [K,N] row-major (B' = B; attention-apply P @ V)
// 256 threads = 8 wave32, wave grid 2 (M) x 4 (N).
// Each wave computes (BM/32) x (BN/64) subtiles of 16x16.
// M,N multiples of BM/BN; K multiple of 32 (true for all calls here).
// ---------------------------------------------------------------------------
template<int BM, int BN, bool BT>
__global__ __launch_bounds__(256) void gemm_f16_wmma(
    const float* __restrict__ A, const float* __restrict__ B,
    const float* __restrict__ bias, float* __restrict__ C,
    int M, int N, int K, float alpha)
{
    constexpr int BK  = 32;
    constexpr int LS  = BK + 8;          // LDS row stride in halves (80B: 16B-aligned, bank-spread)
    constexpr int MI  = BM / 32;         // subtiles per wave (M)
    constexpr int NI  = BN / 64;         // subtiles per wave (N)
    constexpr int AV4 = BM * 8 / 256;    // float4 loads per thread for A tile
    constexpr int B4  = BN / 4;
    constexpr int BV4 = BT ? (BN * 8 / 256) : (32 * B4 / 256);

    __shared__ _Float16 As[2][BM * LS];
    __shared__ _Float16 Bs[2][BN * LS];

    const int t    = threadIdx.x;
    const int lane = t & 31;
    const int wave = t >> 5;
    const int hs   = lane >> 4;          // half-wave select (K-chunk select)
    const int l15  = lane & 15;
    const int wr   = wave >> 2;          // 0..1  wave row
    const int wc   = wave & 3;           // 0..3  wave col

    const int gm = blockIdx.y * BM;
    const int gn = blockIdx.x * BN;

    v8f acc[MI][NI];
#pragma unroll
    for (int i = 0; i < MI; ++i)
#pragma unroll
        for (int j = 0; j < NI; ++j)
#pragma unroll
            for (int e = 0; e < 8; ++e) acc[i][j][e] = 0.0f;

    v4f aReg[AV4], bReg[BV4];

    auto loadA = [&](int kb) {
#pragma unroll
        for (int it = 0; it < AV4; ++it) {
            int idx = t + it * 256;
            int row = idx >> 3, c4 = idx & 7;
            aReg[it] = *(const v4f*)(A + (size_t)(gm + row) * K + kb + c4 * 4);
        }
    };
    auto loadB = [&](int kb) {
        if (BT) {
#pragma unroll
            for (int it = 0; it < BV4; ++it) {
                int idx = t + it * 256;
                int row = idx >> 3, c4 = idx & 7;
                bReg[it] = *(const v4f*)(B + (size_t)(gn + row) * K + kb + c4 * 4);
            }
        } else {
#pragma unroll
            for (int it = 0; it < BV4; ++it) {
                int idx  = t + it * 256;
                int krow = idx / B4, n4 = idx % B4;
                bReg[it] = *(const v4f*)(B + (size_t)(kb + krow) * N + gn + n4 * 4);
            }
        }
    };
    auto stageA = [&](int buf) {
#pragma unroll
        for (int it = 0; it < AV4; ++it) {
            int idx = t + it * 256;
            int row = idx >> 3, c4 = idx & 7;
            v4h h;
#pragma unroll
            for (int e = 0; e < 4; ++e) h[e] = (_Float16)aReg[it][e];
            *(v4h*)(&As[buf][row * LS + c4 * 4]) = h;
        }
    };
    auto stageB = [&](int buf) {
        if (BT) {
#pragma unroll
            for (int it = 0; it < BV4; ++it) {
                int idx = t + it * 256;
                int row = idx >> 3, c4 = idx & 7;
                v4h h;
#pragma unroll
                for (int e = 0; e < 4; ++e) h[e] = (_Float16)bReg[it][e];
                *(v4h*)(&Bs[buf][row * LS + c4 * 4]) = h;
            }
        } else {
#pragma unroll
            for (int it = 0; it < BV4; ++it) {
                int idx  = t + it * 256;
                int krow = idx / B4, n4 = idx % B4;
#pragma unroll
                for (int e = 0; e < 4; ++e)
                    Bs[buf][(n4 * 4 + e) * LS + krow] = (_Float16)bReg[it][e];
            }
        }
    };

    // ---- prologue: fill buffer 0 ----
    loadA(0); loadB(0);
    stageA(0); stageB(0);
    __syncthreads();

    int buf = 0;
    for (int kb = 0; kb < K; kb += BK) {
        const bool more = (kb + BK) < K;
        if (more) { loadA(kb + BK); loadB(kb + BK); }   // prefetch next chunk

        // ---- fragment loads (documented 16-bit wave32 VGPR layout) ----
        v16h af[MI], bf[NI];
#pragma unroll
        for (int i = 0; i < MI; ++i) {
            int mo = wr * (BM / 2) + i * 16 + l15;
            v8h lo = *(const v8h*)(&As[buf][mo * LS + hs * 8]);       // K = 8*hs .. +7
            v8h hi = *(const v8h*)(&As[buf][mo * LS + 16 + hs * 8]);  // K = 16+8*hs .. +7
#pragma unroll
            for (int e = 0; e < 8; ++e) { af[i][e] = lo[e]; af[i][8 + e] = hi[e]; }
        }
#pragma unroll
        for (int j = 0; j < NI; ++j) {
            int no = wc * (BN / 4) + j * 16 + l15;
            v8h lo = *(const v8h*)(&Bs[buf][no * LS + hs * 8]);       // K = 8*hs .. +7
            v8h hi = *(const v8h*)(&Bs[buf][no * LS + 16 + hs * 8]);  // K = 16+8*hs .. +7
#pragma unroll
            for (int e = 0; e < 8; ++e) { bf[j][e] = lo[e]; bf[j][8 + e] = hi[e]; }
        }

#pragma unroll
        for (int i = 0; i < MI; ++i)
#pragma unroll
            for (int j = 0; j < NI; ++j)
                acc[i][j] = __builtin_amdgcn_wmma_f32_16x16x32_f16(
                    false, af[i], false, bf[j], (short)0, acc[i][j], false, false);

        if (more) {
            stageA(buf ^ 1); stageB(buf ^ 1);   // write other buffer (readers use `buf`)
            __syncthreads();
            buf ^= 1;
        }
    }

    // ---- epilogue: C = alpha*acc + bias ----
#pragma unroll
    for (int j = 0; j < NI; ++j) {
        int col = gn + wc * (BN / 4) + j * 16 + l15;
        float bv = bias ? bias[col] : 0.0f;
#pragma unroll
        for (int i = 0; i < MI; ++i) {
#pragma unroll
            for (int r = 0; r < 8; ++r) {
                int row = gm + wr * (BM / 2) + i * 16 + r + 8 * hs;
                C[(size_t)row * N + col] = alpha * acc[i][j][r] + bv;
            }
        }
    }
}

// ---------------------------------------------------------------------------
// Row softmax (in place), one block per row.
// ---------------------------------------------------------------------------
__global__ __launch_bounds__(256) void softmax_rows(float* __restrict__ P, int cols)
{
    __shared__ float red[256];
    float* p = P + (size_t)blockIdx.x * cols;
    int t = threadIdx.x;

    float m = -3.4e38f;
    for (int c = t; c < cols; c += 256) m = fmaxf(m, p[c]);
    red[t] = m; __syncthreads();
    for (int s = 128; s > 0; s >>= 1) { if (t < s) red[t] = fmaxf(red[t], red[t + s]); __syncthreads(); }
    m = red[0]; __syncthreads();

    float sum = 0.0f;
    for (int c = t; c < cols; c += 256) { float e = __expf(p[c] - m); p[c] = e; sum += e; }
    red[t] = sum; __syncthreads();
    for (int s = 128; s > 0; s >>= 1) { if (t < s) red[t] += red[t + s]; __syncthreads(); }
    float inv = 1.0f / red[0];
    for (int c = t; c < cols; c += 256) p[c] *= inv;
}

__global__ __launch_bounds__(256) void add_relu2(
    const float* __restrict__ a, const float* __restrict__ b,
    float* __restrict__ o, int n)
{
    int i = blockIdx.x * 256 + threadIdx.x;
    if (i < n) o[i] = fmaxf(a[i] + b[i], 0.0f);
}

__global__ __launch_bounds__(256) void add_relu3(
    const float* __restrict__ a, const float* __restrict__ b,
    const float* __restrict__ c, float* __restrict__ o, int n)
{
    int i = blockIdx.x * 256 + threadIdx.x;
    if (i < n) o[i] = fmaxf(a[i] + b[i] + c[i], 0.0f);
}

// ---------------------------------------------------------------------------
extern "C" void kernel_launch(void* const* d_in, const int* in_sizes, int n_in,
                              void* d_out, int out_size, void* d_ws, size_t ws_size,
                              hipStream_t stream)
{
    (void)in_sizes; (void)n_in; (void)out_size; (void)ws_size;

    const float* x        = (const float*)d_in[0];   // [512,1024,7,7] == [512,50176] row-major
    const float* fc1_w    = (const float*)d_in[1];
    const float* fc1_b    = (const float*)d_in[2];
    const float* fc2_w    = (const float*)d_in[3];
    const float* fc2_b    = (const float*)d_in[4];
    const float* a1_fc1_w = (const float*)d_in[5];
    const float* a1_fc1_b = (const float*)d_in[6];
    const float* a1_fc2_w = (const float*)d_in[7];
    const float* a1_fc2_b = (const float*)d_in[8];
    const float* a1_cw    = (const float*)d_in[9];
    const float* a1_cb    = (const float*)d_in[10];
    const float* a2_fc1_w = (const float*)d_in[11];
    const float* a2_fc1_b = (const float*)d_in[12];
    const float* a2_fc2_w = (const float*)d_in[13];
    const float* a2_fc2_b = (const float*)d_in[14];
    const float* a2_cw    = (const float*)d_in[15];
    const float* a2_cb    = (const float*)d_in[16];

    const int ND = NPROP * DDIM;         // 524288
    float* out0   = (float*)d_ws;        // [512,1024] fc1 output (live to end)
    float* bufA   = out0   + ND;         // q / h0 scratch
    float* bufB   = bufA   + ND;         // k scratch
    float* logits = bufB   + ND;         // [512,512]
    float* attn   = logits + NPROP * NPROP;
    float* rbuf   = attn   + ND;
    float* hbuf   = rbuf   + ND;         // fc2 output (live to end)
    float* outp   = (float*)d_out;

    dim3 blk(256);
    dim3 gFC1(DDIM / 128, NPROP / 128);          // 8 x 4
    dim3 gDD (DDIM / 64,  NPROP / 64);           // 16 x 8
    dim3 gNN (NPROP / 64, NPROP / 64);           // 8 x 8
    dim3 gEl ((ND + 255) / 256);

    // ---- fc1: out0 = x @ fc1_w^T + b ----
    gemm_f16_wmma<128,128,true><<<gFC1, blk, 0, stream>>>(
        x, fc1_w, fc1_b, out0, NPROP, DDIM, FLATK, 1.0f);

    // ---- attention 1 on out0 ----
    gemm_f16_wmma<64,64,true><<<gDD, blk, 0, stream>>>(
        out0, a1_fc1_w, a1_fc1_b, bufA, NPROP, DDIM, DDIM, 1.0f);       // q
    gemm_f16_wmma<64,64,true><<<gDD, blk, 0, stream>>>(
        out0, a1_fc2_w, a1_fc2_b, bufB, NPROP, DDIM, DDIM, 1.0f);       // k
    gemm_f16_wmma<64,64,true><<<gNN, blk, 0, stream>>>(
        bufA, bufB, nullptr, logits, NPROP, NPROP, DDIM, ATTN_SCALE);   // q@k^T * scale
    softmax_rows<<<NPROP, blk, 0, stream>>>(logits, NPROP);
    gemm_f16_wmma<64,64,false><<<gDD, blk, 0, stream>>>(
        logits, out0, nullptr, attn, NPROP, DDIM, NPROP, 1.0f);         // p @ out0
    gemm_f16_wmma<64,64,true><<<gDD, blk, 0, stream>>>(
        attn, a1_cw, a1_cb, rbuf, NPROP, DDIM, DDIM, 1.0f);             // r1

    // ---- h0 = relu(out0 + r1); h = h0 @ fc2_w^T + b ----
    add_relu2<<<gEl, blk, 0, stream>>>(out0, rbuf, bufA, ND);
    gemm_f16_wmma<64,64,true><<<gDD, blk, 0, stream>>>(
        bufA, fc2_w, fc2_b, hbuf, NPROP, DDIM, DDIM, 1.0f);

    // ---- attention 2 on h ----
    gemm_f16_wmma<64,64,true><<<gDD, blk, 0, stream>>>(
        hbuf, a2_fc1_w, a2_fc1_b, bufA, NPROP, DDIM, DDIM, 1.0f);       // q
    gemm_f16_wmma<64,64,true><<<gDD, blk, 0, stream>>>(
        hbuf, a2_fc2_w, a2_fc2_b, bufB, NPROP, DDIM, DDIM, 1.0f);       // k
    gemm_f16_wmma<64,64,true><<<gNN, blk, 0, stream>>>(
        bufA, bufB, nullptr, logits, NPROP, NPROP, DDIM, ATTN_SCALE);
    softmax_rows<<<NPROP, blk, 0, stream>>>(logits, NPROP);
    gemm_f16_wmma<64,64,false><<<gDD, blk, 0, stream>>>(
        logits, hbuf, nullptr, attn, NPROP, DDIM, NPROP, 1.0f);         // p @ h
    gemm_f16_wmma<64,64,true><<<gDD, blk, 0, stream>>>(
        attn, a2_cw, a2_cb, rbuf, NPROP, DDIM, DDIM, 1.0f);             // r2

    // ---- out = relu(h + r2 + out0) ----
    add_relu3<<<gEl, blk, 0, stream>>>(hbuf, rbuf, out0, outp, ND);
}
